// rnn_11055245820535
// MI455X (gfx1250) — compile-verified, matrix operand order checked by reading, not run on hardware
//
#include <hip/hip_runtime.h>

typedef unsigned short u16;
typedef __attribute__((ext_vector_type(16))) __bf16 v16bf;
typedef __attribute__((ext_vector_type(8)))  float  v8f;

#define E_DIM 512
#define B_SZ  8
#define T_SZ  512
#define V_SZ  32000

#define WMMA_BF16(a, b, c) \
  __builtin_amdgcn_wmma_f32_16x16x32_bf16(false, (a), false, (b), (short)0, (c), false, false)

__device__ __forceinline__ u16 f2bf(float f) {
  unsigned u = __float_as_uint(f);
  u += 0x7FFFu + ((u >> 16) & 1u);
  return (u16)(u >> 16);
}

// ---------------- fp32 -> bf16 conversion (grid-stride) ----------------
__global__ void cvt_f32_bf16(const float* __restrict__ s, u16* __restrict__ d, long n) {
  long i = (long)blockIdx.x * blockDim.x + threadIdx.x;
  long stride = (long)gridDim.x * blockDim.x;
  for (; i < n; i += stride) d[i] = f2bf(s[i]);
}

// ---------------- embedding gather: tok_emb[x] -> bf16 [B*T, E] --------
__global__ void embed_gather(const int* __restrict__ x, const float* __restrict__ tok,
                             u16* __restrict__ xe, long ntok) {
  long i = (long)blockIdx.x * blockDim.x + threadIdx.x;
  long n = ntok * E_DIM;
  long stride = (long)gridDim.x * blockDim.x;
  for (; i < n; i += stride) {
    long t = i / E_DIM;
    int  e = (int)(i - t * E_DIM);
    xe[i] = f2bf(tok[(long)x[t] * E_DIM + e]);
  }
}

// ======================================================================
// Head-path WMMA GEMM: C[M,N] = A[M,K]*W[N,K]^T + bias.  Wave tile 32x64:
// 2 A frags x 4 B frags -> 8 WMMA per 6 fragment loads (B reused over M).
// Double-buffered: next chunk's loads issued before current chunk's WMMAs
// so the WMMA chain runs under partial loadcnt (no s_wait_loadcnt 0).
// M%32==0, N%64==0, K%64==0.
// ======================================================================
__global__ void __launch_bounds__(256)
wmma_gemm_32x64(const u16* __restrict__ A, const u16* __restrict__ W,
                const float* __restrict__ bias, float* __restrict__ Cf,
                u16* __restrict__ Cb, int M, int N, int K, int relu, int lda) {
  const int lane = threadIdx.x & 31;
  const int wave = threadIdx.x >> 5;
  const int Mt = M >> 5;
  const int Nt = N >> 6;
  long tile = (long)blockIdx.x * (blockDim.x >> 5) + wave;
  if (tile >= (long)Mt * Nt) return;            // wave-uniform
  const int tn = (int)(tile % Nt);
  const int tm = (int)(tile / Nt);

  const int  kh    = (lane >> 4) << 4;          // K-half of the 32-chunk
  const u16* arow0 = A + (long)(tm * 32 + (lane & 15)) * lda + kh;
  const u16* arow1 = arow0 + (long)16 * lda;
  const u16* wrow  = W + (long)(tn * 64 + (lane & 15)) * K + kh;
  const long ws16  = (long)16 * K;

  v8f c00 = {}, c01 = {}, c02 = {}, c03 = {};
  v8f c10 = {}, c11 = {}, c12 = {}, c13 = {};

  v16bf a0 = *(const v16bf*)(arow0);
  v16bf a1 = *(const v16bf*)(arow1);
  v16bf b0 = *(const v16bf*)(wrow);
  v16bf b1 = *(const v16bf*)(wrow + ws16);
  v16bf b2 = *(const v16bf*)(wrow + 2 * ws16);
  v16bf b3 = *(const v16bf*)(wrow + 3 * ws16);

  for (int k0 = 32; k0 < K; k0 += 32) {
    v16bf a0n = *(const v16bf*)(arow0 + k0);
    v16bf a1n = *(const v16bf*)(arow1 + k0);
    v16bf b0n = *(const v16bf*)(wrow + k0);
    v16bf b1n = *(const v16bf*)(wrow + ws16 + k0);
    v16bf b2n = *(const v16bf*)(wrow + 2 * ws16 + k0);
    v16bf b3n = *(const v16bf*)(wrow + 3 * ws16 + k0);
    __builtin_prefetch(wrow + k0 + 224, 0, 1);  // keep weight stream ahead in L2
    c00 = WMMA_BF16(a0, b0, c00);
    c10 = WMMA_BF16(a1, b0, c10);
    c01 = WMMA_BF16(a0, b1, c01);
    c11 = WMMA_BF16(a1, b1, c11);
    c02 = WMMA_BF16(a0, b2, c02);
    c12 = WMMA_BF16(a1, b2, c12);
    c03 = WMMA_BF16(a0, b3, c03);
    c13 = WMMA_BF16(a1, b3, c13);
    a0 = a0n; a1 = a1n; b0 = b0n; b1 = b1n; b2 = b2n; b3 = b3n;
  }
  c00 = WMMA_BF16(a0, b0, c00);
  c10 = WMMA_BF16(a1, b0, c10);
  c01 = WMMA_BF16(a0, b1, c01);
  c11 = WMMA_BF16(a1, b1, c11);
  c02 = WMMA_BF16(a0, b2, c02);
  c12 = WMMA_BF16(a1, b2, c12);
  c03 = WMMA_BF16(a0, b3, c03);
  c13 = WMMA_BF16(a1, b3, c13);

  v8f cl[2][4] = {{c00, c01, c02, c03}, {c10, c11, c12, c13}};
  const int mb = tm * 32 + ((lane >> 4) << 3);  // C frag: m = 8*(l>>4)+r, n = l&15
#pragma unroll
  for (int h = 0; h < 2; ++h) {
#pragma unroll
    for (int j = 0; j < 4; ++j) {
      int n = tn * 64 + j * 16 + (lane & 15);
      float bv = bias ? bias[n] : 0.0f;
#pragma unroll
      for (int r = 0; r < 8; ++r) {
        float v = cl[h][j][r] + bv;
        if (relu) v = fmaxf(v, 0.0f);
        long idx = (long)(mb + h * 16 + r) * N + n;
        if (Cf) Cf[idx] = v;
        if (Cb) Cb[idx] = f2bf(v);
      }
    }
  }
}

// ======================================================================
// Split-K WMMA GEMM for the M=16(padded) scan layers: each wave computes a
// 16x64 tile over a K-slice of kc, writing fp32 partials [S][16][N].
// Double-buffered like above.  kc % 32 == 0, kc >= 64.
// ======================================================================
__global__ void __launch_bounds__(256)
wmma_gemm_splitk(const u16* __restrict__ A, const u16* __restrict__ W,
                 float* __restrict__ part, int N, int K, int kc, int lda) {
  const int lane = threadIdx.x & 31;
  const int wave = threadIdx.x >> 5;
  const int Nt = N >> 6;
  const int S  = K / kc;
  long gw = (long)blockIdx.x * (blockDim.x >> 5) + wave;
  if (gw >= (long)Nt * S) return;
  const int tn = (int)(gw % Nt);
  const int s  = (int)(gw / Nt);

  const int  kh   = (lane >> 4) << 4;
  const u16* arow = A + (long)(lane & 15) * lda + s * kc + kh;
  const u16* wrow = W + (long)(tn * 64 + (lane & 15)) * K + s * kc + kh;
  const long ws16 = (long)16 * K;

  v8f c0 = {}, c1 = {}, c2 = {}, c3 = {};
  v16bf a  = *(const v16bf*)(arow);
  v16bf b0 = *(const v16bf*)(wrow);
  v16bf b1 = *(const v16bf*)(wrow + ws16);
  v16bf b2 = *(const v16bf*)(wrow + 2 * ws16);
  v16bf b3 = *(const v16bf*)(wrow + 3 * ws16);

  for (int k0 = 32; k0 < kc; k0 += 32) {
    v16bf an  = *(const v16bf*)(arow + k0);
    v16bf b0n = *(const v16bf*)(wrow + k0);
    v16bf b1n = *(const v16bf*)(wrow + ws16 + k0);
    v16bf b2n = *(const v16bf*)(wrow + 2 * ws16 + k0);
    v16bf b3n = *(const v16bf*)(wrow + 3 * ws16 + k0);
    __builtin_prefetch(wrow + k0 + 224, 0, 1);
    c0 = WMMA_BF16(a, b0, c0);
    c1 = WMMA_BF16(a, b1, c1);
    c2 = WMMA_BF16(a, b2, c2);
    c3 = WMMA_BF16(a, b3, c3);
    a = an; b0 = b0n; b1 = b1n; b2 = b2n; b3 = b3n;
  }
  c0 = WMMA_BF16(a, b0, c0);
  c1 = WMMA_BF16(a, b1, c1);
  c2 = WMMA_BF16(a, b2, c2);
  c3 = WMMA_BF16(a, b3, c3);

  float* pb = part + (long)s * 16 * N;
  const int mb = (lane >> 4) << 3;
  v8f acc[4] = {c0, c1, c2, c3};
#pragma unroll
  for (int j = 0; j < 4; ++j) {
    int n = tn * 64 + j * 16 + (lane & 15);
#pragma unroll
    for (int r = 0; r < 8; ++r) pb[(long)(mb + r) * N + n] = acc[j][r];
  }
}

// ---- sum split-K partials + bias + ReLU + cvt, over a [16, N] tile -----
__global__ void reduce_bias_act(const float* __restrict__ part, int S,
                                const float* __restrict__ bias, float* __restrict__ Cf,
                                u16* __restrict__ Cb, int N, int relu) {
  int i = blockIdx.x * blockDim.x + threadIdx.x;
  if (i >= 16 * N) return;
  int n = i % N;
  float v = bias[n];
  for (int s = 0; s < S; ++s) v += part[(long)s * 16 * N + i];
  if (relu) v = fmaxf(v, 0.0f);
  if (Cf) Cf[i] = v;
  if (Cb) Cb[i] = f2bf(v);
}

// ---- x0 tile for beg GEMM: padded [16, E], rows 8..15 zero -------------
__global__ void pack_x0(const u16* __restrict__ xe, u16* __restrict__ x0) {
  int i = blockIdx.x * blockDim.x + threadIdx.x;   // 16*512
  if (i < 16 * E_DIM) {
    int r = i >> 9, e = i & 511;
    x0[i] = (r < B_SZ) ? xe[((long)r * T_SZ) * E_DIM + e] : (u16)0;
  }
}

// ---- recurrent-step input concat: [xt, carry[:, :E]], padded 16 rows ---
__global__ void build_inp(const u16* __restrict__ xe, int t, const u16* __restrict__ carry,
                          u16* __restrict__ inp) {
  int i = blockIdx.x * blockDim.x + threadIdx.x;   // 16*1024
  if (i < 16 * 1024) {
    int b = i >> 10, j = i & 1023;
    if (b < B_SZ)
      inp[i] = (j < E_DIM) ? xe[((long)b * T_SZ + t) * E_DIM + j]
                           : carry[b * 1024 + (j - E_DIM)];
    else
      inp[i] = 0;   // keep pad rows deterministic / NaN-free
  }
}

// ---------------- LayerNorm over 1024 + carry/result writeback ---------
__global__ void ln_step(const float* __restrict__ h, const float* __restrict__ g,
                        const float* __restrict__ bta, u16* __restrict__ carry_bf,
                        float* __restrict__ res, int t) {
  const int b = blockIdx.x;                        // 0..7
  const int tid = threadIdx.x;                     // 256
  __shared__ float s1[256], s2[256];
  const float* hr = h + b * 1024;
  float sum = 0.f, ss = 0.f;
  for (int j = tid; j < 1024; j += 256) { float v = hr[j]; sum += v; ss += v * v; }
  s1[tid] = sum; s2[tid] = ss;
  __syncthreads();
  for (int w = 128; w > 0; w >>= 1) {
    if (tid < w) { s1[tid] += s1[tid + w]; s2[tid] += s2[tid + w]; }
    __syncthreads();
  }
  float mu  = s1[0] * (1.0f / 1024.0f);
  float var = s2[0] * (1.0f / 1024.0f) - mu * mu;
  float inv = rsqrtf(var + 1e-5f);
  for (int j = tid; j < 1024; j += 256) {
    float v = (hr[j] - mu) * inv * g[j] + bta[j];
    carry_bf[b * 1024 + j] = f2bf(v);
    if (j >= E_DIM) res[((long)b * T_SZ + t) * E_DIM + (j - E_DIM)] = v;
  }
}

// ---------------- r0 = x_s0[:, E:] -------------------------------------
__global__ void copy_r0(const float* __restrict__ s0, float* __restrict__ res) {
  int i = blockIdx.x * blockDim.x + threadIdx.x;   // 8*512
  if (i < B_SZ * E_DIM) {
    int b = i >> 9, e = i & 511;
    res[((long)b * T_SZ) * E_DIM + e] = s0[b * 1024 + E_DIM + e];
  }
}

extern "C" void kernel_launch(void* const* d_in, const int* in_sizes, int n_in,
                              void* d_out, int out_size, void* d_ws, size_t ws_size,
                              hipStream_t stream) {
  (void)in_sizes; (void)n_in; (void)out_size; (void)ws_size;
  const int*   x     = (const int*)d_in[0];
  const float* tok   = (const float*)d_in[1];
  const float* beg_w = (const float*)d_in[2];
  const float* beg_b = (const float*)d_in[3];
  const float* w1 = (const float*)d_in[4];  const float* b1 = (const float*)d_in[5];
  const float* w2 = (const float*)d_in[6];  const float* b2 = (const float*)d_in[7];
  const float* w3 = (const float*)d_in[8];  const float* b3 = (const float*)d_in[9];
  const float* w4 = (const float*)d_in[10]; const float* b4 = (const float*)d_in[11];
  const float* w5 = (const float*)d_in[12]; const float* b5 = (const float*)d_in[13];
  const float* ln_g = (const float*)d_in[14]; const float* ln_b = (const float*)d_in[15];
  const float* hw1 = (const float*)d_in[16]; const float* hb1 = (const float*)d_in[17];
  const float* hw2 = (const float*)d_in[18]; const float* hb2 = (const float*)d_in[19];
  const float* hw3 = (const float*)d_in[20]; const float* hb3 = (const float*)d_in[21];

  float* logits = (float*)d_out;
  float* result = logits + (long)B_SZ * T_SZ * V_SZ;

  char* ws = (char*)d_ws; size_t off = 0;
  auto alloc = [&](size_t bytes) -> void* {
    void* p = ws + off; off += (bytes + 255) & ~(size_t)255; return p;
  };
  // bf16 weight mirrors (converted once; scan weights stay resident in L2)
  u16* beg_w_bf = (u16*)alloc(2L * 1024 * 512);
  u16* w1_bf  = (u16*)alloc(2L * 2048 * 1024);
  u16* w2_bf  = (u16*)alloc(2L * 4096 * 2048);
  u16* w3_bf  = (u16*)alloc(2L * 4096 * 4096);
  u16* w4_bf  = (u16*)alloc(2L * 2048 * 4096);
  u16* w5_bf  = (u16*)alloc(2L * 1024 * 2048);
  u16* hw1_bf = (u16*)alloc(2L * 1024 * 512);
  u16* hw2_bf = (u16*)alloc(2L * 2048 * 1024);
  u16* hw3_bf = (u16*)alloc(2L * (long)V_SZ * 2048);
  // activations (scan buffers padded to 16 rows)
  u16* xe_bf  = (u16*)alloc(2L * B_SZ * T_SZ * E_DIM);
  u16* hh1_bf = (u16*)alloc(2L * 4096 * 1024);
  u16* hh2_bf = (u16*)alloc(2L * 4096 * 2048);
  u16* x0_bf  = (u16*)alloc(2L * 16 * E_DIM);
  u16* inp_bf = (u16*)alloc(2L * 16 * 1024);
  u16* h1_bf  = (u16*)alloc(2L * 16 * 2048);
  u16* h2_bf  = (u16*)alloc(2L * 16 * 4096);
  u16* h3_bf  = (u16*)alloc(2L * 16 * 4096);
  u16* h4_bf  = (u16*)alloc(2L * 16 * 2048);
  float* h5_f = (float*)alloc(4L * 16 * 1024);
  float* s0_f = (float*)alloc(4L * 16 * 1024);
  u16* carry_bf = (u16*)alloc(2L * 16 * 1024);
  float* part = (float*)alloc(4L * 16 * 16 * 4096);   // split-K partials (max 4 MB)

  auto cvt = [&](const float* s, u16* d, long n) {
    long b = (n + 255) / 256; if (b > 16384) b = 16384;
    cvt_f32_bf16<<<(int)b, 256, 0, stream>>>(s, d, n);
  };
  auto gemm = [&](const u16* A, const u16* Wt, const float* bias, float* Cf, u16* Cb,
                  int M, int N, int K, int relu, int lda) {
    long tiles = (long)(M >> 5) * (N >> 6);
    wmma_gemm_32x64<<<(int)((tiles + 7) / 8), 256, 0, stream>>>(A, Wt, bias, Cf, Cb,
                                                                M, N, K, relu, lda);
  };
  auto rgemm = [&](const u16* A, const u16* Wt, const float* bias, float* Cf, u16* Cb,
                   int N, int K, int relu, int lda) {
    const int kc = 256;                    // K-slice per wave (8 WMMA chunks)
    const int S  = K / kc;
    long waves = (long)(N >> 6) * S;
    wmma_gemm_splitk<<<(int)((waves + 7) / 8), 256, 0, stream>>>(A, Wt, part, N, K, kc, lda);
    int tot = 16 * N;
    reduce_bias_act<<<(tot + 255) / 256, 256, 0, stream>>>(part, S, bias, Cf, Cb, N, relu);
  };

  // ---- one-time conversions ----
  cvt(beg_w, beg_w_bf, 1024L * 512);
  cvt(w1, w1_bf, 2048L * 1024);
  cvt(w2, w2_bf, 4096L * 2048);
  cvt(w3, w3_bf, 4096L * 4096);
  cvt(w4, w4_bf, 2048L * 4096);
  cvt(w5, w5_bf, 1024L * 2048);
  cvt(hw1, hw1_bf, 1024L * 512);
  cvt(hw2, hw2_bf, 2048L * 1024);
  cvt(hw3, hw3_bf, (long)V_SZ * 2048);
  embed_gather<<<8192, 256, 0, stream>>>(x, tok, xe_bf, (long)B_SZ * T_SZ);

  // ---- head path: bulk WMMA GEMMs, 32x64 wave tiles ----
  gemm(xe_bf,  hw1_bf, hb1, nullptr, hh1_bf, B_SZ * T_SZ, 1024, 512,  1, 512);
  gemm(hh1_bf, hw2_bf, hb2, nullptr, hh2_bf, B_SZ * T_SZ, 2048, 1024, 1, 1024);
  gemm(hh2_bf, hw3_bf, hb3, logits,  nullptr, B_SZ * T_SZ, V_SZ, 2048, 0, 2048);

  // ---- recurrent init: x_s0 = xe[:,0] @ beg_w^T + beg_b ----
  pack_x0<<<32, 256, 0, stream>>>(xe_bf, x0_bf);
  rgemm(x0_bf, beg_w_bf, beg_b, s0_f, carry_bf, 1024, 512, 0, 512);
  copy_r0<<<16, 256, 0, stream>>>(s0_f, result);

  // ---- sequential scan: 511 steps; split-K widens weight streaming ----
  for (int t = 1; t < T_SZ; ++t) {
    build_inp<<<64, 256, 0, stream>>>(xe_bf, t, carry_bf, inp_bf);
    rgemm(inp_bf, w1_bf, b1, nullptr, h1_bf, 2048, 1024, 1, 1024);
    rgemm(h1_bf,  w2_bf, b2, nullptr, h2_bf, 4096, 2048, 1, 2048);
    rgemm(h2_bf,  w3_bf, b3, nullptr, h3_bf, 4096, 4096, 1, 4096);
    rgemm(h3_bf,  w4_bf, b4, nullptr, h4_bf, 2048, 4096, 1, 4096);
    rgemm(h4_bf,  w5_bf, b5, h5_f,    nullptr, 1024, 2048, 1, 2048);
    ln_step<<<B_SZ, 256, 0, stream>>>(h5_f, ln_g, ln_b, carry_bf, result, t);
  }
}
